// BotRGCN_5531917877296
// MI455X (gfx1250) — compile-verified, matrix operand order checked by reading, not run on hardware
//
#include <hip/hip_runtime.h>
#include <hip/hip_bf16.h>

typedef __attribute__((ext_vector_type(16))) _Float16 v16h;
typedef __attribute__((ext_vector_type(8)))  float    v8f;

#define D 128

// ---------------------------------------------------------------------------
// Encoder: per node, 4 small linears (fan-in 100/100/6/11 -> 32 each) + lrelu,
// concatenated into x0[N,128]. One block (128 threads) per node; inputs staged
// in LDS so each row is read once from HBM.
// ---------------------------------------------------------------------------
__global__ void bot_encoder(const float* __restrict__ des,
                            const float* __restrict__ tweet,
                            const float* __restrict__ nump,
                            const float* __restrict__ catp,
                            const float* __restrict__ Wd, const float* __restrict__ bd,
                            const float* __restrict__ Wt, const float* __restrict__ bt,
                            const float* __restrict__ Wn, const float* __restrict__ bn,
                            const float* __restrict__ Wc, const float* __restrict__ bc,
                            float* __restrict__ x0, int nnodes) {
    __shared__ float s[220];
    const int n = blockIdx.x;
    if (n >= nnodes) return;
    const int j = threadIdx.x;            // 0..127
    if (j < 100) s[j]       = des  [(size_t)n * 100 + j];
    if (j < 100) s[100 + j] = tweet[(size_t)n * 100 + j];
    if (j < 6)   s[200 + j] = nump [(size_t)n * 6   + j];
    if (j < 11)  s[206 + j] = catp [(size_t)n * 11  + j];
    __syncthreads();

    const int part = j >> 5;              // which 32-wide output slice
    const int u    = j & 31;
    float acc;
    if (part == 0) {
        acc = bd[u];
        #pragma unroll 4
        for (int i = 0; i < 100; ++i) acc += s[i] * Wd[i * 32 + u];
    } else if (part == 1) {
        acc = bt[u];
        #pragma unroll 4
        for (int i = 0; i < 100; ++i) acc += s[100 + i] * Wt[i * 32 + u];
    } else if (part == 2) {
        acc = bn[u];
        #pragma unroll
        for (int i = 0; i < 6; ++i)   acc += s[200 + i] * Wn[i * 32 + u];
    } else {
        acc = bc[u];
        #pragma unroll
        for (int i = 0; i < 11; ++i)  acc += s[206 + i] * Wc[i * 32 + u];
    }
    acc = (acc > 0.0f) ? acc : 0.01f * acc;          // leaky relu
    x0[(size_t)n * D + j] = acc;
}

// ---------------------------------------------------------------------------
// Pack a 128x128 f32 weight matrix into WMMA B-fragment layout, f16.
// Record layout: out[(((ct*4 + kb4)*32 + lane)*16) + 2i .. +1] where
//   k = kb4*32 + 16*(i/4) + 8*(lane/16) + 2*(i%4), col = ct*16 + lane%16.
// Each lane's 16-half fragment is one contiguous 32-byte span.
// 8192 threads total (one per f16 pair).
// ---------------------------------------------------------------------------
__global__ void pack_w16(const float* __restrict__ W, _Float16* __restrict__ out) {
    const int t    = blockIdx.x * blockDim.x + threadIdx.x;   // 0..8191
    const int i    = t & 7;
    const int lane = (t >> 3) & 31;
    const int kb4  = (t >> 8) & 3;
    const int ct   = t >> 10;                                  // 0..7
    const int half = lane >> 4;
    const int l16  = lane & 15;
    const int k    = kb4 * 32 + 16 * (i >> 2) + 8 * half + 2 * (i & 3);
    const int col  = ct * 16 + l16;
    _Float16* rec = out + (size_t)(((ct * 4 + kb4) * 32 + lane) * 16);
    rec[2 * i]     = (_Float16)W[(size_t)k * D + col];
    rec[2 * i + 1] = (_Float16)W[(size_t)(k + 1) * D + col];
}

// ---------------------------------------------------------------------------
// WMMA GEMM: C[nrows,128] = act(A[nrows,128] @ W[128,128] (+ bias))
// One wave per 16-row stripe covering ALL 8 column tiles:
//   - A row fragments loaded as contiguous b128 spans, converted once to f16
//     (4x v16h = 32 VGPRs), then reused across the 8 column tiles.
//   - B fragments (packed f16, L2-resident) reloaded per column tile.
//   - 32x v_wmma_f32_16x16x32_f16 per wave; A read exactly once per GEMM.
// Row clamp avoids load divergence; store guard is wave-uniform.
// ---------------------------------------------------------------------------
__global__ void gemm128_wmma(const float* __restrict__ A,
                             const _Float16* __restrict__ Bpk,
                             const float* __restrict__ bias,
                             float* __restrict__ C,
                             int nrows, int act) {
    const int lane = threadIdx.x & 31;
    const int wave = threadIdx.x >> 5;                 // 0..3
    const int rowTile = blockIdx.x * 4 + wave;
    const int row0 = rowTile * 16;
    if (row0 >= nrows) return;                         // wave-uniform

    const int half = lane >> 4;                        // 0 or 1
    const int l16  = lane & 15;
    int r = row0 + l16;
    r = (r < nrows) ? r : (nrows - 1);                 // clamp: no divergence
    const float4* ap = (const float4*)(A + (size_t)r * D);

    // Load + convert the full A row stripe once: a[kb4] covers K=[32*kb4,32*kb4+32)
    v16h a[4];
    #pragma unroll
    for (int kb4 = 0; kb4 < 4; ++kb4) {
        // Lane needs K spans [kb+8h, kb+8h+8) and [kb+16+8h, ...+8)
        const float4 c0 = ap[kb4 * 8 + 2 * half];
        const float4 c1 = ap[kb4 * 8 + 2 * half + 1];
        const float4 c2 = ap[kb4 * 8 + 4 + 2 * half];
        const float4 c3 = ap[kb4 * 8 + 5 + 2 * half];
        a[kb4][0]  = (_Float16)c0.x;  a[kb4][1]  = (_Float16)c0.y;
        a[kb4][2]  = (_Float16)c0.z;  a[kb4][3]  = (_Float16)c0.w;
        a[kb4][4]  = (_Float16)c1.x;  a[kb4][5]  = (_Float16)c1.y;
        a[kb4][6]  = (_Float16)c1.z;  a[kb4][7]  = (_Float16)c1.w;
        a[kb4][8]  = (_Float16)c2.x;  a[kb4][9]  = (_Float16)c2.y;
        a[kb4][10] = (_Float16)c2.z;  a[kb4][11] = (_Float16)c2.w;
        a[kb4][12] = (_Float16)c3.x;  a[kb4][13] = (_Float16)c3.y;
        a[kb4][14] = (_Float16)c3.z;  a[kb4][15] = (_Float16)c3.w;
    }

    const bool full = (row0 + 16 <= nrows);            // wave-uniform

    #pragma unroll
    for (int ct = 0; ct < 8; ++ct) {
        const _Float16* brec = Bpk + (size_t)(((ct * 4) * 32 + lane) * 16);
        v8f acc = {};
        #pragma unroll
        for (int kb4 = 0; kb4 < 4; ++kb4) {
            const v16h b = *(const v16h*)(brec + (size_t)kb4 * 32 * 16);
            acc = __builtin_amdgcn_wmma_f32_16x16x32_f16(
                false, a[kb4], false, b, (short)0, acc, false, false);
        }

        const int ncol = ct * 16 + l16;
        const float bv = bias ? bias[ncol] : 0.0f;
        float* cp = C + (size_t)(row0 + 8 * half) * D + ncol;
        if (full) {
            #pragma unroll
            for (int j = 0; j < 8; ++j) {
                float v = acc[j] + bv;
                if (act) v = (v > 0.0f) ? v : 0.01f * v;
                cp[(size_t)j * D] = v;
            }
        } else {
            #pragma unroll
            for (int j = 0; j < 8; ++j) {
                const int m = row0 + j + 8 * half;
                if (m < nrows) {
                    float v = acc[j] + bv;
                    if (act) v = (v > 0.0f) ? v : 0.01f * v;
                    C[(size_t)m * D + ncol] = v;
                }
            }
        }
    }
}

// ---------------------------------------------------------------------------
// Zero fill (sums / counts between layers; ws is poisoned by the harness).
// ---------------------------------------------------------------------------
__global__ void zero_f32(float* __restrict__ p, size_t n) {
    size_t i = (size_t)blockIdx.x * blockDim.x + threadIdx.x;
    if (i < n) p[i] = 0.0f;
}

// ---------------------------------------------------------------------------
// Edge scatter: one wave per edge. Gather h[etype, src, :] (b128 per lane)
// and atomically accumulate into sums[(dst*R+etype), :]; count per segment.
// ---------------------------------------------------------------------------
__global__ void edge_scatter(const int* __restrict__ ei,   // [2,E]
                             const int* __restrict__ et,   // [E]
                             const float* __restrict__ h,  // [R,N,128]
                             float* __restrict__ sums,     // [N*R,128]
                             float* __restrict__ cnt,      // [N*R]
                             int nedges, int nnodes) {
    const int wid  = (int)(((size_t)blockIdx.x * blockDim.x + threadIdx.x) >> 5);
    const int lane = threadIdx.x & 31;
    if (wid >= nedges) return;
    const int src = ei[wid];
    const int dst = ei[nedges + wid];
    const int t   = et[wid];

    const float* hrow = h    + ((size_t)t * nnodes + src) * D;
    float*       srow = sums + ((size_t)dst * 2 + t) * D;
    __builtin_prefetch(srow, 1, 1);                     // global_prefetch_b8

    const float4 v = *reinterpret_cast<const float4*>(hrow + lane * 4);
    atomicAdd(&srow[lane * 4 + 0], v.x);
    atomicAdd(&srow[lane * 4 + 1], v.y);
    atomicAdd(&srow[lane * 4 + 2], v.z);
    atomicAdd(&srow[lane * 4 + 3], v.w);
    if (lane == 0) atomicAdd(&cnt[(size_t)dst * 2 + t], 1.0f);
}

// ---------------------------------------------------------------------------
// Combine: x_next[n,f] (holding x@W_root) += mean_r0 + mean_r1 + b_rgcn[f]
// ---------------------------------------------------------------------------
__global__ void rgcn_combine(float* __restrict__ xnext,
                             const float* __restrict__ sums,
                             const float* __restrict__ cnt,
                             const float* __restrict__ brg,
                             int nnodes) {
    const size_t idx = (size_t)blockIdx.x * blockDim.x + threadIdx.x;
    if (idx >= (size_t)nnodes * D) return;
    const size_t n = idx / D;
    const int    f = (int)(idx % D);
    const float c0 = fmaxf(cnt[n * 2 + 0], 1.0f);
    const float c1 = fmaxf(cnt[n * 2 + 1], 1.0f);
    const float m0 = sums[(n * 2 + 0) * D + f] / c0;
    const float m1 = sums[(n * 2 + 1) * D + f] / c1;
    xnext[idx] = xnext[idx] + m0 + m1 + brg[f];
}

// ---------------------------------------------------------------------------
// Head: out[n,0:2] = x[n,:] @ W_o2[128,2] + b_o2. One thread per node.
// ---------------------------------------------------------------------------
__global__ void head2(const float* __restrict__ x,
                      const float* __restrict__ W2,
                      const float* __restrict__ b2,
                      float* __restrict__ out, int nnodes) {
    const int n = (int)((size_t)blockIdx.x * blockDim.x + threadIdx.x);
    if (n >= nnodes) return;
    float a0 = b2[0], a1 = b2[1];
    const float* xr = x + (size_t)n * D;
    #pragma unroll 8
    for (int k = 0; k < D; ++k) {
        const float xv = xr[k];
        a0 += xv * W2[k * 2 + 0];
        a1 += xv * W2[k * 2 + 1];
    }
    out[(size_t)n * 2 + 0] = a0;
    out[(size_t)n * 2 + 1] = a1;
}

// ---------------------------------------------------------------------------
// Host-side orchestration
// ---------------------------------------------------------------------------
extern "C" void kernel_launch(void* const* d_in, const int* in_sizes, int n_in,
                              void* d_out, int out_size, void* d_ws, size_t ws_size,
                              hipStream_t stream) {
    (void)n_in; (void)out_size; (void)ws_size;
    const float* des   = (const float*)d_in[0];
    const float* tweet = (const float*)d_in[1];
    const float* nump  = (const float*)d_in[2];
    const float* catp  = (const float*)d_in[3];
    const int*   ei    = (const int*)d_in[4];
    const int*   et    = (const int*)d_in[5];
    const float* Wd    = (const float*)d_in[6];
    const float* bd    = (const float*)d_in[7];
    const float* Wt    = (const float*)d_in[8];
    const float* bt    = (const float*)d_in[9];
    const float* Wn    = (const float*)d_in[10];
    const float* bn    = (const float*)d_in[11];
    const float* Wc    = (const float*)d_in[12];
    const float* bc    = (const float*)d_in[13];
    const float* Win   = (const float*)d_in[14];
    const float* bin   = (const float*)d_in[15];
    const float* Wrel  = (const float*)d_in[16];   // [2,128,128]
    const float* Wroot = (const float*)d_in[17];
    const float* brg   = (const float*)d_in[18];
    const float* Wo1   = (const float*)d_in[19];
    const float* bo1   = (const float*)d_in[20];
    const float* Wo2   = (const float*)d_in[21];
    const float* bo2   = (const float*)d_in[22];
    float* out = (float*)d_out;

    const int nnodes = in_sizes[0] / 100;          // N
    const int nedges = in_sizes[5];                // E
    const size_t ND = (size_t)nnodes * D;
    const size_t WPK = 128 * 128;                  // halves per packed matrix

    // Workspace layout: packed f16 weights first (32B-aligned records), floats after.
    _Float16* wpk   = (_Float16*)d_ws;             // 5 * 16384 halves
    _Float16* pWin  = wpk + 0 * WPK;
    _Float16* pRel0 = wpk + 1 * WPK;
    _Float16* pRel1 = wpk + 2 * WPK;
    _Float16* pRoot = wpk + 3 * WPK;
    _Float16* pWo1  = wpk + 4 * WPK;

    float* fbase = (float*)(wpk + 5 * WPK);
    float* x0    = fbase;               // 128N  (encoder output, free after W_in)
    float* xcur  = x0   + ND;           // 128N
    float* h     = xcur + ND;           // 256N  (relation-transformed, [R,N,128])
    float* xnext = h    + 2 * ND;       // 128N
    float* sums  = xnext + ND;          // 256N
    float* cnt   = sums + 2 * ND;       // 2N

    const int nRowTiles = (nnodes + 15) / 16;
    const dim3 gblk(128);                           // 4 waves, 1 row stripe each
    const dim3 ggrd((nRowTiles + 3) / 4);

    // 0) pack all five 128x128 weight matrices to f16 B-fragment layout
    pack_w16<<<32, 256, 0, stream>>>(Win,           pWin);
    pack_w16<<<32, 256, 0, stream>>>(Wrel,          pRel0);
    pack_w16<<<32, 256, 0, stream>>>(Wrel + D * D,  pRel1);
    pack_w16<<<32, 256, 0, stream>>>(Wroot,         pRoot);
    pack_w16<<<32, 256, 0, stream>>>(Wo1,           pWo1);

    // 1) feature encoders -> x0
    bot_encoder<<<nnodes, 128, 0, stream>>>(des, tweet, nump, catp,
                                            Wd, bd, Wt, bt, Wn, bn, Wc, bc,
                                            x0, nnodes);
    // 2) x = lrelu(x0 @ W_in + b_in)
    gemm128_wmma<<<ggrd, gblk, 0, stream>>>(x0, pWin, bin, xcur, nnodes, 1);

    // 3) two RGCN layers
    for (int layer = 0; layer < 2; ++layer) {
        // relation transforms h_r = x @ W_rel[r]
        gemm128_wmma<<<ggrd, gblk, 0, stream>>>(xcur, pRel0, nullptr, h,      nnodes, 0);
        gemm128_wmma<<<ggrd, gblk, 0, stream>>>(xcur, pRel1, nullptr, h + ND, nnodes, 0);
        // zero segment accumulators (sums + cnt contiguous)
        {
            const size_t ztot = 2 * ND + 2 * (size_t)nnodes;
            zero_f32<<<(unsigned)((ztot + 255) / 256), 256, 0, stream>>>(sums, ztot);
        }
        // edge gather + scatter-add (wave per edge)
        {
            const unsigned blocks = (unsigned)(((size_t)nedges * 32 + 255) / 256);
            edge_scatter<<<blocks, 256, 0, stream>>>(ei, et, h, sums, cnt, nedges, nnodes);
        }
        // root transform into xnext, then add mean-aggregated messages + bias
        gemm128_wmma<<<ggrd, gblk, 0, stream>>>(xcur, pRoot, nullptr, xnext, nnodes, 0);
        rgcn_combine<<<(unsigned)((ND + 255) / 256), 256, 0, stream>>>(xnext, sums, cnt, brg, nnodes);
        // swap
        float* tswap = xcur; xcur = xnext; xnext = tswap;
    }

    // 4) x = lrelu(x @ W_o1 + b_o1)
    gemm128_wmma<<<ggrd, gblk, 0, stream>>>(xcur, pWo1, bo1, xnext, nnodes, 1);

    // 5) out = x @ W_o2 + b_o2
    head2<<<(unsigned)((nnodes + 255) / 256), 256, 0, stream>>>(xnext, Wo2, bo2, out, nnodes);
}
